// GAT_81011673137280
// MI455X (gfx1250) — matrix operand order repaired and checked by reading, not compiled, not run on hardware
//
#include <hip/hip_runtime.h>

#define N_NODES 50000
#define N_EDGES 800000
#define DIM_IN  256
#define DIM_HID 128
#define DIM_OUT 64

typedef __attribute__((ext_vector_type(16))) __bf16 v16bf;
typedef __attribute__((ext_vector_type(8)))  float  v8f;

__device__ __forceinline__ unsigned short f2bf(float f) {
  unsigned u = __float_as_uint(f);
  u += 0x7FFFu + ((u >> 16) & 1u);          // round-to-nearest-even
  return (unsigned short)(u >> 16);
}

__device__ __forceinline__ float sigmoidf_(float x) {
  return 1.0f / (1.0f + __expf(-x));
}

// ---------------- degree / normalization ----------------
__global__ void k_init_deg(float* __restrict__ deg) {
  int i = blockIdx.x * blockDim.x + threadIdx.x;
  if (i < N_NODES) deg[i] = 1.0f;           // self loop
}

__global__ void k_deg_edges(const int* __restrict__ dst, float* __restrict__ deg) {
  int e = blockIdx.x * blockDim.x + threadIdx.x;
  if (e < N_EDGES) atomicAdd(&deg[dst[e]], 1.0f);
}

__global__ void k_dinv(const float* __restrict__ deg, float* __restrict__ dinv) {
  int i = blockIdx.x * blockDim.x + threadIdx.x;
  if (i < N_NODES) dinv[i] = rsqrtf(deg[i]);
}

// ---------------- precision conversion ----------------
__global__ void k_cast_bf16(const float* __restrict__ src,
                            unsigned short* __restrict__ dst, int count) {
  int i = blockIdx.x * blockDim.x + threadIdx.x;
  if (i < count) dst[i] = f2bf(src[i]);
}

template<int K, int NC>
__global__ void k_transpose_cast(const float* __restrict__ W,       // [K,NC] f32
                                 unsigned short* __restrict__ Wt) { // [NC,K] bf16
  int i = blockIdx.x * blockDim.x + threadIdx.x;
  if (i < K * NC) {
    int k = i / NC, n = i % NC;
    Wt[n * K + k] = f2bf(W[i]);
  }
}

// ---------------- tiled bf16 WMMA GEMM: C[M,NC] = A[M,K] * Bt[NC,K]^T ----------------
// block = 16 rows x NC cols; one wave per 16x16 tile; Bt staged in LDS.
template<int K, int NC>
__global__ __launch_bounds__(32 * (NC / 16))
void k_gemm_wmma(const unsigned short* __restrict__ A,   // [M,K] bf16 row-major
                 const unsigned short* __restrict__ Bt,  // [NC,K] bf16 (transposed weights)
                 float* __restrict__ C) {                // [M,NC] f32
  __shared__ __align__(16) unsigned short ldsB[NC * K];

  const int tid = threadIdx.x;
  {
    const uint4* s = (const uint4*)Bt;
    uint4* d = (uint4*)ldsB;
    const int nvec = NC * K / 8;             // 8 bf16 per uint4
    for (int i = tid; i < nvec; i += blockDim.x) d[i] = s[i];
  }
  __syncthreads();

  const int wave = tid >> 5;
  const int lane = tid & 31;
  const int hi   = lane >> 4;                // half-wave select
  const int n    = lane & 15;                // tile row (A) / tile col (B,C)
  const int colBase = wave << 4;
  const long rowBase = (long)blockIdx.x << 4;

  const unsigned short* Arow = A + (rowBase + n) * K;
  const unsigned short* Brow = ldsB + (colBase + n) * K;
  const int kA = hi << 3;                    // A half-wave K offset: 0 / 8
  const int kB = hi << 3;                    // B half-wave K offset: 0 / 8

  __builtin_prefetch(Arow, 0, 1);            // global_prefetch_b8

  v8f acc = {};
  #pragma unroll
  for (int kk = 0; kk < K; kk += 32) {
    union { v16bf v; uint4 u[2]; } a, b;
    // ISA 16-bit A 16x32 layout: lanes 0-15 hold K 0-7 & 16-23; lanes 16-31 K 8-15 & 24-31
    a.u[0] = *(const uint4*)(Arow + kk + kA);
    a.u[1] = *(const uint4*)(Arow + kk + kA + 16);
    // ISA 16-bit B 32x16 layout: lanes 0-15 hold K 0-7 & 16-23; lanes 16-31 K 8-15 & 24-31
    b.u[0] = *(const uint4*)(Brow + kk + kB);
    b.u[1] = *(const uint4*)(Brow + kk + kB + 16);
    acc = __builtin_amdgcn_wmma_f32_16x16x32_bf16(
        false, a.v, false, b.v, (short)0, acc, false, false);
  }

  // C/D layout: VGPR r -> M = r + 8*hi, N = lane&15
  float* Cp = C + rowBase * NC + colBase + n;
  #pragma unroll
  for (int r = 0; r < 8; ++r)
    Cp[(long)(r + (hi << 3)) * NC] = acc[r];
}

// ---------------- graph aggregation ----------------
template<int C>
__global__ void k_self_init(const float* __restrict__ h, const float* __restrict__ dinv,
                            float* __restrict__ agg) {
  long i = (long)blockIdx.x * blockDim.x + threadIdx.x;
  if (i < (long)N_NODES * C) {
    int node = (int)(i / C);
    float d = dinv[node];
    agg[i] = d * d * h[i];                   // self-loop term, norm = dinv^2
  }
}

template<int C>
__global__ void k_scatter(const int* __restrict__ src, const int* __restrict__ dst,
                          const float* __restrict__ dinv, const float* __restrict__ h,
                          float* __restrict__ agg) {
  const int EPB = 256 / C;                   // edges per block
  int e = blockIdx.x * EPB + threadIdx.x / C;
  if (e < N_EDGES) {
    int c = threadIdx.x % C;
    int s = src[e], d = dst[e];
    float w = dinv[s] * dinv[d];
    atomicAdd(&agg[(long)d * C + c], w * h[(long)s * C + c]);
  }
}

// ---------------- epilogues: bias (+relu) + sigmoid gate ----------------
__global__ void k_epilogue1(const float* __restrict__ agg, const float* __restrict__ b1,
                            const float* __restrict__ aw1, const float* __restrict__ ab1,
                            unsigned short* __restrict__ h1b) {
  int node = blockIdx.x * 8 + (threadIdx.x >> 5);
  int lane = threadIdx.x & 31;
  const float* row = agg + (long)node * DIM_HID;
  float v[4];
  float partial = 0.f;
  #pragma unroll
  for (int i = 0; i < 4; ++i) {
    int c = lane + 32 * i;
    float t = fmaxf(row[c] + b1[c], 0.f);    // relu BEFORE gate
    v[i] = t;
    partial += t * aw1[c];
  }
  #pragma unroll
  for (int m = 16; m >= 1; m >>= 1) partial += __shfl_xor(partial, m, 32);
  float g = sigmoidf_(partial + ab1[0]);
  unsigned short* out = h1b + (long)node * DIM_HID;
  #pragma unroll
  for (int i = 0; i < 4; ++i) out[lane + 32 * i] = f2bf(v[i] * g);
}

__global__ void k_epilogue2(const float* __restrict__ agg, const float* __restrict__ b2,
                            const float* __restrict__ aw2, const float* __restrict__ ab2,
                            float* __restrict__ out) {
  int node = blockIdx.x * 8 + (threadIdx.x >> 5);
  int lane = threadIdx.x & 31;
  const float* row = agg + (long)node * DIM_OUT;
  float v[2];
  float partial = 0.f;
  #pragma unroll
  for (int i = 0; i < 2; ++i) {
    int c = lane + 32 * i;
    float t = row[c] + b2[c];                // no relu on second conv
    v[i] = t;
    partial += t * aw2[c];
  }
  #pragma unroll
  for (int m = 16; m >= 1; m >>= 1) partial += __shfl_xor(partial, m, 32);
  float g = sigmoidf_(partial + ab2[0]);
  float* o = out + (long)node * DIM_OUT;
  #pragma unroll
  for (int i = 0; i < 2; ++i) o[lane + 32 * i] = v[i] * g;
}

// ---------------- launch ----------------
extern "C" void kernel_launch(void* const* d_in, const int* in_sizes, int n_in,
                              void* d_out, int out_size, void* d_ws, size_t ws_size,
                              hipStream_t stream) {
  const float* x   = (const float*)d_in[0];
  const int*   ei  = (const int*)d_in[1];   // edge_index [2,E]
  const float* W1  = (const float*)d_in[2];
  const float* b1  = (const float*)d_in[3];
  const float* W2  = (const float*)d_in[4];
  const float* b2  = (const float*)d_in[5];
  const float* aw1 = (const float*)d_in[6];
  const float* ab1 = (const float*)d_in[7];
  const float* aw2 = (const float*)d_in[8];
  const float* ab2 = (const float*)d_in[9];
  const int* src = ei;
  const int* dst = ei + N_EDGES;

  char* ws = (char*)d_ws;
  size_t off = 0;
  auto alloc = [&](size_t bytes) {
    char* p = ws + off;
    off += (bytes + 255) & ~(size_t)255;
    return p;
  };
  float*          deg  = (float*)alloc((size_t)N_NODES * 4);
  float*          dinv = (float*)alloc((size_t)N_NODES * 4);
  unsigned short* xb   = (unsigned short*)alloc((size_t)N_NODES * DIM_IN * 2);
  unsigned short* w1t  = (unsigned short*)alloc((size_t)DIM_IN * DIM_HID * 2);
  unsigned short* w2t  = (unsigned short*)alloc((size_t)DIM_HID * DIM_OUT * 2);
  float*          h1   = (float*)alloc((size_t)N_NODES * DIM_HID * 4);
  float*          agg1 = (float*)alloc((size_t)N_NODES * DIM_HID * 4);
  unsigned short* h1b  = (unsigned short*)alloc((size_t)N_NODES * DIM_HID * 2);
  float*          h2   = (float*)xb;   // reuse (xb dead after GEMM1)
  float*          agg2 = (float*)h1;   // reuse (h1 dead after scatter1)

  // degrees + symmetric normalization
  k_init_deg<<<(N_NODES + 255) / 256, 256, 0, stream>>>(deg);
  k_deg_edges<<<(N_EDGES + 255) / 256, 256, 0, stream>>>(dst, deg);
  k_dinv<<<(N_NODES + 255) / 256, 256, 0, stream>>>(deg, dinv);

  // bf16 conversions
  k_cast_bf16<<<(N_NODES * DIM_IN + 255) / 256, 256, 0, stream>>>(x, xb, N_NODES * DIM_IN);
  k_transpose_cast<DIM_IN, DIM_HID><<<(DIM_IN * DIM_HID + 255) / 256, 256, 0, stream>>>(W1, w1t);
  k_transpose_cast<DIM_HID, DIM_OUT><<<(DIM_HID * DIM_OUT + 255) / 256, 256, 0, stream>>>(W2, w2t);

  // layer 1: GEMM -> aggregate -> bias/relu/gate -> bf16
  k_gemm_wmma<DIM_IN, DIM_HID><<<N_NODES / 16, 32 * (DIM_HID / 16), 0, stream>>>(xb, w1t, h1);
  k_self_init<DIM_HID><<<(int)(((long)N_NODES * DIM_HID + 255) / 256), 256, 0, stream>>>(h1, dinv, agg1);
  k_scatter<DIM_HID><<<(N_EDGES + 1) / 2, 256, 0, stream>>>(src, dst, dinv, h1, agg1);
  k_epilogue1<<<N_NODES / 8, 256, 0, stream>>>(agg1, b1, aw1, ab1, h1b);

  // layer 2: GEMM -> aggregate -> bias/gate -> f32 out
  k_gemm_wmma<DIM_HID, DIM_OUT><<<N_NODES / 16, 32 * (DIM_OUT / 16), 0, stream>>>(h1b, w2t, h2);
  k_self_init<DIM_OUT><<<(int)(((long)N_NODES * DIM_OUT + 255) / 256), 256, 0, stream>>>(h2, dinv, agg2);
  k_scatter<DIM_OUT><<<(N_EDGES + 3) / 4, 256, 0, stream>>>(src, dst, dinv, h2, agg2);
  k_epilogue2<<<N_NODES / 8, 256, 0, stream>>>(agg2, b2, aw2, ab2, (float*)d_out);

  (void)in_sizes; (void)n_in; (void)out_size; (void)ws_size;
}
